// ASPM_13700945674777
// MI455X (gfx1250) — matrix operand order repaired and checked by reading, not compiled, not run on hardware
//
#include <hip/hip_runtime.h>
#include <stdint.h>

#define DD      1024
#define TT      4096
#define BATCH   32
#define ROWS    (BATCH * TT)      // 131072
#define WPB     8                 // waves per block
#define ROWS_PB (WPB * 16)        // 128 rows per block
#define NTILES  (DD / 16)         // 64
#define KSTEPS  (DD / 32)         // 32
#define NMASK   (TT / 2)          // 2048
#define BSTRIDE 1032              // staged B column stride in bf16 (bank spread)
#define BUFELEM (16 * BSTRIDE)    // bf16 elems per B buffer (33024 B)

typedef __bf16 bf16_t;
typedef bf16_t v16bf __attribute__((ext_vector_type(16)));
typedef float  v8f   __attribute__((ext_vector_type(8)));

union Frag16 { uint4 q[2]; v16bf v; };
union Acc8   { v8f v; float f[8]; };

__device__ __forceinline__ uint32_t pk(float a, float b) {
  return (__float_as_uint(a) >> 16) | (__float_as_uint(b) & 0xFFFF0000u);
}

// ---------------------------------------------------------------------------
// Kernel 0: W1 (f32, [k][n] row-major) -> W1p (bf16, [n][k] K-contiguous).
// Makes each 16-column B tile one contiguous 32KB chunk (async-copy friendly).
// ---------------------------------------------------------------------------
__global__ void w1_convert_kernel(const float* __restrict__ W1,
                                  uint16_t* __restrict__ W1p) {
  __shared__ float tile[32][33];
  const int tx = threadIdx.x, ty = threadIdx.y;
  const int n0 = blockIdx.x * 32, k0 = blockIdx.y * 32;
  tile[ty][tx] = W1[(size_t)(k0 + ty) * DD + (n0 + tx)];
  __syncthreads();
  const float v = tile[tx][ty];  // == W1[k0+tx][n0+ty]
  W1p[(size_t)(n0 + ty) * DD + (k0 + tx)] =
      (uint16_t)(__float_as_uint(v) >> 16);
}

// ---------------------------------------------------------------------------
// Kernel 1: fused  scores = tanh(x @ W1 + b1) @ w2 + b2   (h never hits HBM)
// 8 waves/block, one 16-row tile per wave. A register-resident (256 VGPRs).
// B tiles double-buffered in LDS via global_load_async_to_lds_b128.
// ---------------------------------------------------------------------------
__device__ __forceinline__ void stage_b_async(const uint16_t* __restrict__ W1p,
                                              uint16_t* bufBase, int nt, int tid) {
#pragma unroll
  for (int j = 0; j < 8; ++j) {
    const int i16  = j * 256 + tid;   // uint4 index within 32KB tile, 0..2047
    const int col  = i16 >> 7;        // 0..15  (128 uint4 per column)
    const int koff = i16 & 127;
    const uint16_t* g = W1p + (size_t)(nt * 16 + col) * DD + koff * 8;
    const uint32_t lds = (uint32_t)(uintptr_t)(bufBase + col * BSTRIDE + koff * 8);
    asm volatile("global_load_async_to_lds_b128 %0, %1, off"
                 :: "v"(lds), "v"(g) : "memory");
  }
}

__global__ __launch_bounds__(WPB * 32)
void fused_score_kernel(const float* __restrict__ x,
                        const uint16_t* __restrict__ W1p,  // bf16 [n*DD + k]
                        const float* __restrict__ b1,
                        const float* __restrict__ w2,
                        const float* __restrict__ b2,
                        float* __restrict__ scores) {
  extern __shared__ uint16_t sB[];  // 2 * BUFELEM bf16 (double buffer)
  const int tid     = threadIdx.x;
  const int lane    = tid & 31;
  const int wv      = tid >> 5;
  const int half    = lane >> 4;    // 0: lanes 0-15, 1: lanes 16-31
  const int mrow    = lane & 15;
  const int rowBase = blockIdx.x * ROWS_PB + wv * 16;

  // --- A: load this wave's 16x1024 x tile straight into registers as bf16 ---
  // Each element is read by exactly one lane; fragments held across all nt.
  Frag16 afrag[KSTEPS];             // 32 * 32B = 256 VGPRs
  {
    const float* xrow = x + (size_t)(rowBase + mrow) * DD;
#pragma unroll
    for (int ks = 0; ks < KSTEPS; ++ks) {
      const int kb = ks * 32;
      const float4* p0 = (const float4*)(xrow + kb + half * 8);
      const float4* p1 = (const float4*)(xrow + kb + 16 + half * 8);
      float4 fa = p0[0], fb = p0[1], fc = p1[0], fd = p1[1];
      afrag[ks].q[0] = make_uint4(pk(fa.x, fa.y), pk(fa.z, fa.w),
                                  pk(fb.x, fb.y), pk(fb.z, fb.w));
      afrag[ks].q[1] = make_uint4(pk(fc.x, fc.y), pk(fc.z, fc.w),
                                  pk(fd.x, fd.y), pk(fd.z, fd.w));
    }
  }

  // --- prefetch B tile 0 into buffer 0 ---
  stage_b_async(W1p, sB, 0, tid);

  float part[8];
#pragma unroll
  for (int r = 0; r < 8; ++r) part[r] = 0.0f;

  for (int nt = 0; nt < NTILES; ++nt) {
    uint16_t* cur = sB + (nt & 1) * BUFELEM;
    // prefetch next tile into the other buffer, then wait for current tile
    if (nt + 1 < NTILES) {
      stage_b_async(W1p, sB + ((nt + 1) & 1) * BUFELEM, nt + 1, tid);
      asm volatile("s_wait_asynccnt 0x8" ::: "memory");  // tile nt complete
    } else {
      asm volatile("s_wait_asynccnt 0x0" ::: "memory");
    }
    __syncthreads();  // all waves' portions of tile nt visible

    const uint16_t* cbuf = cur + mrow * BSTRIDE;  // column n_local = mrow
    Acc8 acc;
#pragma unroll
    for (int r = 0; r < 8; ++r) acc.f[r] = 0.0f;

#pragma unroll
    for (int ks = 0; ks < KSTEPS; ++ks) {
      const int kb = ks * 32;
      // B fragment (32x16 bf16): lanes 0-15 K 0..7,16..23, lanes 16-31 K 8..15,24..31
      Frag16 bf;
      const uint16_t* bp = cbuf + kb + half * 8;
      bf.q[0] = *(const uint4*)bp;
      bf.q[1] = *(const uint4*)(bp + 16);
      acc.v = __builtin_amdgcn_wmma_f32_16x16x32_bf16(
          false, afrag[ks].v, false, bf.v, (short)0, acc.v, false, false);
    }

    // epilogue on 16x16 tile: lane holds rows r+(half?8:0), col n
    const int n = nt * 16 + mrow;
    const float bias = b1[n];
    const float w2n  = w2[n];
#pragma unroll
    for (int r = 0; r < 8; ++r) {
      const float h = tanhf(acc.f[r] + bias);
      part[r] += h * w2n;
    }
    __syncthreads();  // everyone done reading 'cur' before it is overwritten
  }

  // reduce across the 16 lanes of each half (same M, different N)
#pragma unroll
  for (int r = 0; r < 8; ++r) {
    float v = part[r];
    v += __shfl_xor(v, 1, 32);
    v += __shfl_xor(v, 2, 32);
    v += __shfl_xor(v, 4, 32);
    v += __shfl_xor(v, 8, 32);
    part[r] = v;
  }
  if ((lane & 15) == 0) {
    const float bb2 = b2[0];
    const int base = rowBase + (half ? 8 : 0);
#pragma unroll
    for (int r = 0; r < 8; ++r) scores[base + r] = part[r] + bb2;
  }
}

// ---------------------------------------------------------------------------
// Kernel 2: per batch row — mask 2048 smallest scores (stable ties), softmax.
// ---------------------------------------------------------------------------
__device__ __forceinline__ int blockReduceSumI(int v, int* sred, int lane, int wv) {
  v += __shfl_xor(v, 16, 32);
  v += __shfl_xor(v, 8, 32);
  v += __shfl_xor(v, 4, 32);
  v += __shfl_xor(v, 2, 32);
  v += __shfl_xor(v, 1, 32);
  if (lane == 0) sred[wv] = v;
  __syncthreads();
  if (threadIdx.x == 0) {
    int t = 0;
    for (int j = 0; j < 8; ++j) t += sred[j];
    sred[0] = t;
  }
  __syncthreads();
  const int total = sred[0];
  __syncthreads();
  return total;
}

__device__ __forceinline__ float blockReduceF(float v, float* sred, int lane,
                                              int wv, bool doMax) {
  for (int m = 16; m >= 1; m >>= 1) {
    float o = __shfl_xor(v, m, 32);
    v = doMax ? fmaxf(v, o) : (v + o);
  }
  if (lane == 0) sred[wv] = v;
  __syncthreads();
  if (threadIdx.x == 0) {
    float t = sred[0];
    for (int j = 1; j < 8; ++j) t = doMax ? fmaxf(t, sred[j]) : (t + sred[j]);
    sred[0] = t;
  }
  __syncthreads();
  const float total = sred[0];
  __syncthreads();
  return total;
}

__global__ __launch_bounds__(256)
void select_softmax_kernel(const float* __restrict__ scores,
                           float* __restrict__ weights) {
  __shared__ uint32_t skey[TT];
  __shared__ float    sval[TT];
  __shared__ int      sredi[8];
  __shared__ float    sredf[8];

  const int b = blockIdx.x;
  const int tid = threadIdx.x;
  const int lane = tid & 31, wv = tid >> 5;
  const float* srow = scores + (size_t)b * TT;

  for (int i = tid; i < TT; i += 256) {
    const float f = srow[i];
    uint32_t u = __float_as_uint(f);
    u = (u & 0x80000000u) ? ~u : (u | 0x80000000u);  // order-preserving key
    skey[i] = u;
    sval[i] = f;
  }
  __syncthreads();

  // binary search for v = NMASK-th smallest key (minimal key with count(<=v) >= NMASK)
  uint32_t lo = 0u, hi = 0xFFFFFFFFu;
  for (int it = 0; it < 32; ++it) {
    const uint32_t mid = (uint32_t)(((uint64_t)lo + (uint64_t)hi) >> 1);
    int cnt = 0;
    for (int i = tid; i < TT; i += 256) cnt += (skey[i] <= mid) ? 1 : 0;
    const int total = blockReduceSumI(cnt, sredi, lane, wv);
    if (total >= NMASK) hi = mid; else lo = mid + 1u;
  }
  const uint32_t v = hi;

  int cl = 0;
  for (int i = tid; i < TT; i += 256) cl += (skey[i] < v) ? 1 : 0;
  const int count_less = blockReduceSumI(cl, sredi, lane, wv);
  const int rneed = NMASK - count_less;  // equals-to-v to mask (lowest indices)

  bool kept[16];
  float lmax = -INFINITY;
#pragma unroll
  for (int e = 0; e < 16; ++e) {
    const int i = tid * 16 + e;
    const uint32_t k = skey[i];
    bool masked;
    if (k < v) {
      masked = true;
    } else if (k == v) {
      int rank = 0;
      for (int j = 0; j < i; ++j) rank += (skey[j] == v) ? 1 : 0;
      masked = (rank < rneed);   // stable argsort: lowest indices masked first
    } else {
      masked = false;
    }
    kept[e] = !masked;
    if (!masked) lmax = fmaxf(lmax, sval[i]);
  }

  const float gmax = blockReduceF(lmax, sredf, lane, wv, true);

  float lsum = 0.0f;
  float ex[16];
#pragma unroll
  for (int e = 0; e < 16; ++e) {
    const int i = tid * 16 + e;
    const float t = kept[e] ? __expf(sval[i] - gmax) : 0.0f;
    ex[e] = t;
    lsum += t;
  }
  const float gsum = blockReduceF(lsum, sredf, lane, wv, false);
  const float inv = 1.0f / gsum;

  float* wrow = weights + (size_t)b * TT;
#pragma unroll
  for (int e = 0; e < 16; ++e) {
    const int i = tid * 16 + e;
    wrow[i] = ex[e] * inv;
  }
}

// ---------------------------------------------------------------------------
// Kernel 3: masked_output = x * weights[row]   (pure streaming, float4)
// ---------------------------------------------------------------------------
__global__ __launch_bounds__(256)
void scale_kernel(const float* __restrict__ x,
                  const float* __restrict__ weights,
                  float* __restrict__ out) {
  const int row = blockIdx.x;
  const float wv = weights[row];
  const float4* xr = (const float4*)(x + (size_t)row * DD);
  float4* orow = (float4*)(out + (size_t)row * DD);
  const int i = threadIdx.x;  // 256 threads * float4 == 1024 floats
  float4 f = xr[i];
  f.x *= wv; f.y *= wv; f.z *= wv; f.w *= wv;
  orow[i] = f;
}

// ---------------------------------------------------------------------------
extern "C" void kernel_launch(void* const* d_in, const int* in_sizes, int n_in,
                              void* d_out, int out_size, void* d_ws, size_t ws_size,
                              hipStream_t stream) {
  (void)in_sizes; (void)n_in; (void)out_size; (void)ws_size;
  const float* x  = (const float*)d_in[0];
  const float* W1 = (const float*)d_in[1];
  const float* b1 = (const float*)d_in[2];
  const float* w2 = (const float*)d_in[3];
  const float* b2 = (const float*)d_in[4];

  float* out     = (float*)d_out;
  float* weights = out + (size_t)ROWS * DD;   // second output region; also
                                              // used as scratch for raw scores
  uint16_t* W1p  = (uint16_t*)d_ws;           // 1024*1024 bf16 = 2 MB

  // 0) convert + transpose W1 -> bf16 K-contiguous (one-time, L2-resident)
  w1_convert_kernel<<<dim3(DD / 32, DD / 32), dim3(32, 32), 0, stream>>>(W1, W1p);

  // 1) fused GEMM(tanh)->GEMV: raw scores into the weights region
  const size_t ldsBytes = (size_t)2 * BUFELEM * sizeof(uint16_t);  // 66048 B
  fused_score_kernel<<<ROWS / ROWS_PB, WPB * 32, ldsBytes, stream>>>(
      x, W1p, b1, w2, b2, weights);

  // 2) per-row mask-bottom-half + softmax (in place: scores -> weights)
  select_softmax_kernel<<<BATCH, 256, 0, stream>>>(weights, weights);

  // 3) out = x * weights
  scale_kernel<<<ROWS, 256, 0, stream>>>(x, weights, out);
}
